// SimpleProNetVAEModel_65137474011586
// MI455X (gfx1250) — compile-verified, matrix operand order checked by reading, not run on hardware
//
#include <hip/hip_runtime.h>

typedef __attribute__((ext_vector_type(16))) _Float16 v16h;
typedef __attribute__((ext_vector_type(8)))  _Float16 v8h;
typedef __attribute__((ext_vector_type(8)))  float    v8f;

#define HD 256        // hidden dim
#define ED 256        // embedding dim
#define NEC 512       // codebook size
#define NB 64         // graphs
#define NMAXP 1280    // padded length
#define NATOMS 65536  // total atoms
#define MROWS (NB * NMAXP)  // 81920 dense rows

// ---------------------------------------------------------------------------
// WMMA fragment loaders (wave32, V_WMMA_F32_16X16X32_F16 layouts, ISA 7.12.2)
// A 16x32 f16: lane l (l<16): row M=l, halves K = kb..kb+7 then kb+16..kb+23
//              lane l (>=16): same rows, kb = k0+8 (caller passes lg)
// ---------------------------------------------------------------------------
__device__ inline v16h load_a_frag(const _Float16* __restrict__ row, int kb) {
  const v8h* p0 = (const v8h*)(row + kb);
  const v8h* p1 = (const v8h*)(row + kb + 16);
  v8h lo = *p0, hi = *p1;
  return __builtin_shufflevector(lo, hi, 0, 1, 2, 3, 4, 5, 6, 7, 8, 9, 10, 11,
                                 12, 13, 14, 15);
}

// B fragments are pre-packed so each lane's 16 halves are contiguous:
// element index = (((c*2 + lg) * N + n) * 16 + j)
__device__ inline v16h load_bp_frag(const _Float16* __restrict__ Wp, int N,
                                    int c, int lg, int n) {
  const v8h* p = (const v8h*)(Wp + ((size_t)((c * 2 + lg) * N + n) << 4));
  v8h lo = p[0], hi = p[1];
  return __builtin_shufflevector(lo, hi, 0, 1, 2, 3, 4, 5, 6, 7, 8, 9, 10, 11,
                                 12, 13, 14, 15);
}

// ---------------------------------------------------------------------------
// f16-in / f32-accum GEMM: out = act(A[MxK] @ W[KxN] + bias)
// ACT: 0 none, 1 relu, 2 clip(-10,10). One wave -> 32x64 tile (2 M x 4 N),
// so each B fragment feeds two independent WMMA chains and each A fragment
// feeds four; the two accumulator chains give the scheduler independent work.
// ---------------------------------------------------------------------------
template <int ACT, bool W16, bool W32>
__global__ __launch_bounds__(32) void gemm16_kernel(
    const _Float16* __restrict__ A, const _Float16* __restrict__ Wp,
    const float* __restrict__ bias, int N, int K,
    _Float16* __restrict__ out16, float* __restrict__ out32) {
  int lane = threadIdx.x & 31;
  int l15 = lane & 15;
  int lg = lane >> 4;
  int mt = blockIdx.x * 32;
  int nt0 = blockIdx.y * 64;

  v8f acc[2][4];
#pragma unroll
  for (int u = 0; u < 4; ++u) {
    float bv = bias[nt0 + u * 16 + l15];
#pragma unroll
    for (int r = 0; r < 8; ++r) { acc[0][u][r] = bv; acc[1][u][r] = bv; }
  }

  const _Float16* arow0 = A + (size_t)(mt + l15) * K;
  const _Float16* arow1 = arow0 + (size_t)16 * K;
  for (int k0 = 0; k0 < K; k0 += 32) {
    int c = k0 >> 5;
    int kb = k0 + lg * 8;
    v16h a0 = load_a_frag(arow0, kb);
    v16h a1 = load_a_frag(arow1, kb);
    v16h b[4];
#pragma unroll
    for (int u = 0; u < 4; ++u)
      b[u] = load_bp_frag(Wp, N, c, lg, nt0 + u * 16 + l15);
    if (k0 + 32 < K)  // pull next chunk's fragments toward the caches
      __builtin_prefetch(Wp + ((size_t)((c + 1) * 2 + lg) * N + nt0 + l15) * 16,
                         0, 3);
#pragma unroll
    for (int u = 0; u < 4; ++u) {
      acc[0][u] = __builtin_amdgcn_wmma_f32_16x16x32_f16(false, a0, false, b[u],
                                                         (short)0, acc[0][u],
                                                         false, false);
      acc[1][u] = __builtin_amdgcn_wmma_f32_16x16x32_f16(false, a1, false, b[u],
                                                         (short)0, acc[1][u],
                                                         false, false);
    }
  }

#pragma unroll
  for (int mi = 0; mi < 2; ++mi) {
#pragma unroll
    for (int u = 0; u < 4; ++u) {
#pragma unroll
      for (int r = 0; r < 8; ++r) {
        float v = acc[mi][u][r];
        if (ACT == 1) v = fmaxf(v, 0.f);
        if (ACT == 2) v = fminf(10.f, fmaxf(-10.f, v));
        size_t row = (size_t)(mt + mi * 16 + r + lg * 8);
        size_t col = (size_t)(nt0 + u * 16 + l15);
        if constexpr (W32) out32[row * N + col] = v;
        if constexpr (W16) out16[row * N + col] = (_Float16)v;
      }
    }
  }
}

// ---------------------------------------------------------------------------
// VQ: fused distance-GEMM (WMMA) + argmin + masked loss/histogram + gather.
// CBp = -2 * codebook^T pre-packed into B-fragment layout; enorm[n]=||e_n||^2.
// dist(m,n) = ||z_m||^2 + (enorm[n] + z_m . (-2 e_n))  ==  ||z_m - e_n||^2
// One wave handles 16 rows against all 512 codes; A tile register-resident,
// B fragments double-buffered: next tile's loads issue before this tile's
// WMMA chain so their latency hides under the matrix math. All min/argmin
// updates are in select form so they lower to v_cndmask (no EXEC churn).
// ---------------------------------------------------------------------------
__global__ __launch_bounds__(32) void vq_kernel(
    const _Float16* __restrict__ Z, const _Float16* __restrict__ CBp,
    const float* __restrict__ enorm, const float* __restrict__ codebook,
    const float* __restrict__ maskf, float* __restrict__ qout,
    _Float16* __restrict__ q16, float* __restrict__ idx_out_f,
    float* __restrict__ sumsq, float* __restrict__ counts) {
  __shared__ int s_code[16];
  int lane = threadIdx.x & 31;
  int l15 = lane & 15;
  int lg = lane >> 4;
  int mt = blockIdx.x * 16;

  const _Float16* zrow = Z + (size_t)(mt + l15) * ED;
  v16h afr[8];
  float z2p = 0.f;
#pragma unroll
  for (int c = 0; c < 8; ++c) {
    afr[c] = load_a_frag(zrow, c * 32 + lg * 8);
#pragma unroll
    for (int j = 0; j < 16; ++j) { float f = (float)afr[c][j]; z2p += f * f; }
  }
  float z2 = z2p + __shfl_xor(z2p, 16, 32);  // lane l: ||z||^2 of row l15

  float best[8];
  int bidx[8];
#pragma unroll
  for (int r = 0; r < 8; ++r) { best[r] = 3.4e38f; bidx[r] = 0; }

  v16h ba[8], bn[8];
#pragma unroll
  for (int c = 0; c < 8; ++c) ba[c] = load_bp_frag(CBp, NEC, c, lg, l15);

  auto step = [&](v16h (&bc)[8], v16h (&bnx)[8], int nt, bool loadNext) {
    int n = nt * 16 + l15;
    if (loadNext) {  // issue next tile's loads before this tile's WMMAs
#pragma unroll
      for (int c = 0; c < 8; ++c)
        bnx[c] = load_bp_frag(CBp, NEC, c, lg, n + 16);
    }
    float e2 = enorm[n];
    v8f acc;
#pragma unroll
    for (int r = 0; r < 8; ++r) acc[r] = e2;
#pragma unroll
    for (int c = 0; c < 8; ++c)
      acc = __builtin_amdgcn_wmma_f32_16x16x32_f16(false, afr[c], false, bc[c],
                                                   (short)0, acc, false, false);
#pragma unroll
    for (int r = 0; r < 8; ++r) {  // ascending n => strict < keeps first min
      float v = acc[r];
      bool lt = v < best[r];
      best[r] = lt ? v : best[r];
      bidx[r] = lt ? n : bidx[r];
    }
  };

#pragma unroll 1
  for (int nt = 0; nt < NEC / 16; nt += 2) {
    step(ba, bn, nt, true);
    step(bn, ba, nt + 1, nt + 2 < NEC / 16);
  }

  // xor-reduce min across the 16-lane half; every lane ends with the min.
#pragma unroll
  for (int r = 0; r < 8; ++r) {
    float bv = best[r];
    int bi = bidx[r];
#pragma unroll
    for (int off = 8; off >= 1; off >>= 1) {
      float ov = __shfl_xor(bv, off, 32);
      int oi = __shfl_xor(bi, off, 32);
      bool take = (ov < bv) || (ov == bv && oi < bi);
      bv = take ? ov : bv;
      bi = take ? oi : bi;
    }
    best[r] = bv;
    bidx[r] = bi;
  }

  // Row j owned by lane l15==j within the half that computed it (lg = j>>3).
  int j = l15;
  bool writer = (lg == 0) ? (j < 8) : (j >= 8);
  int r = (lg == 0) ? j : (j - 8);
  if (writer) {
    int row = mt + j;
    int code = bidx[r];
    float dist = best[r] + z2;
    if (dist < 0.f) dist = 0.f;
    s_code[j] = code;
    idx_out_f[row] = (float)code;
    float mk = maskf[row];
    if (mk != 0.f) {
      atomicAdd(sumsq, mk * dist);
      atomicAdd(&counts[code], mk);
    }
  }
  __syncthreads();

  // Cooperative gather: quantized = codebook[idx] (f32 output + f16 to decoder)
  for (int rr = 0; rr < 16; ++rr) {
    int code = s_code[rr];
    const float* e = codebook + (size_t)code * ED;
    size_t base = (size_t)(mt + rr) * ED;
    for (int c = lane; c < ED; c += 32) {
      float v = e[c];
      qout[base + c] = v;
      q16[base + c] = (_Float16)v;
    }
  }
}

// ---------------------------------------------------------------------------
// Small helper kernels
// ---------------------------------------------------------------------------
__global__ void scatter_kernel(const float* __restrict__ coords,
                               const int* __restrict__ batch,
                               float* __restrict__ xdense,
                               float* __restrict__ maskf, int n) {
  int i = blockIdx.x * blockDim.x + threadIdx.x;
  if (i >= n) return;
  int b = batch[i];
  int lo = 0, hi = n;  // lower_bound == searchsorted(b, b) (sorted input)
  while (lo < hi) {
    int mid = (lo + hi) >> 1;
    if (batch[mid] < b) lo = mid + 1; else hi = mid;
  }
  int pos = i - lo;
  if (pos < NMAXP) {
    size_t row = (size_t)b * NMAXP + pos;
    xdense[row * 3 + 0] = coords[(size_t)i * 3 + 0];
    xdense[row * 3 + 1] = coords[(size_t)i * 3 + 1];
    xdense[row * 3 + 2] = coords[(size_t)i * 3 + 2];
    maskf[row] = 1.0f;
  }
}

__global__ void enc_l0_kernel(const float* __restrict__ x,
                              const float* __restrict__ w,
                              const float* __restrict__ b,
                              _Float16* __restrict__ out16) {
  int t = blockIdx.x * blockDim.x + threadIdx.x;  // MROWS * HD
  int m = t >> 8, n = t & 255;
  float v = b[n] + x[(size_t)m * 3 + 0] * w[0 * HD + n] +
            x[(size_t)m * 3 + 1] * w[1 * HD + n] +
            x[(size_t)m * 3 + 2] * w[2 * HD + n];
  out16[(size_t)t] = (_Float16)fmaxf(v, 0.f);
}

__global__ void dec_out_kernel(const _Float16* __restrict__ h,
                               const float* __restrict__ w,
                               const float* __restrict__ b,
                               float* __restrict__ out) {
  int t = blockIdx.x * blockDim.x + threadIdx.x;  // MROWS * 3
  if (t >= MROWS * 3) return;
  int m = t / 3, c = t - m * 3;
  const _Float16* hr = h + (size_t)m * HD;
  float acc = b[c];
#pragma unroll 8
  for (int k = 0; k < HD; ++k) acc += (float)hr[k] * w[k * 3 + c];
  out[t] = acc;
}

// Pack f32 matrix into WMMA B-fragment layout (f16, optional scale).
// src element (k, n) at src[k*sk + n*sn];  out[(((c*2+lg)*N + n)*16 + j)]
// with k = c*32 + lg*8 + (j<8 ? j : 8+j).
__global__ void pack_wfrag_kernel(const float* __restrict__ src,
                                  _Float16* __restrict__ out, int N, int K,
                                  int sk, int sn, float scale) {
  int t = blockIdx.x * blockDim.x + threadIdx.x;
  if (t >= K * N) return;
  int j = t & 15;
  int rest = t >> 4;
  int n = rest % N;
  int cl = rest / N;  // c*2 + lg
  int lg = cl & 1, c = cl >> 1;
  int k = c * 32 + lg * 8 + (j < 8 ? j : 8 + j);
  out[t] = (_Float16)(scale * src[(size_t)k * sk + (size_t)n * sn]);
}

__global__ void enorm_kernel(const float* __restrict__ cb,
                             float* __restrict__ enorm) {
  int n = blockIdx.x * blockDim.x + threadIdx.x;
  if (n >= NEC) return;
  const float* e = cb + (size_t)n * ED;
  float s = 0.f;
  for (int k = 0; k < ED; ++k) s += e[k] * e[k];
  enorm[n] = s;
}

__global__ void finalize_kernel(const float* __restrict__ counts,
                                const float* __restrict__ sumsq,
                                float* __restrict__ loss_out,
                                float* __restrict__ perp_out) {
  __shared__ float sh[NEC];
  int t = threadIdx.x;
  float p = counts[t] / (float)NATOMS;
  sh[t] = p * logf(p + 1e-10f);
  __syncthreads();
  for (int s = NEC / 2; s > 0; s >>= 1) {
    if (t < s) sh[t] += sh[t + s];
    __syncthreads();
  }
  if (t == 0) {
    *perp_out = expf(-sh[0]);
    *loss_out = 1.25f * (*sumsq) / ((float)NATOMS * (float)ED);
  }
}

// ---------------------------------------------------------------------------
extern "C" void kernel_launch(void* const* d_in, const int* in_sizes, int n_in,
                              void* d_out, int out_size, void* d_ws,
                              size_t ws_size, hipStream_t stream) {
  (void)in_sizes; (void)n_in; (void)out_size; (void)ws_size;

  const float* coords = (const float*)d_in[0];
  const int* batch = (const int*)d_in[1];
  const float* encW[5]; const float* encB[5];
  for (int i = 0; i < 5; ++i) {
    encW[i] = (const float*)d_in[2 + 2 * i];
    encB[i] = (const float*)d_in[3 + 2 * i];
  }
  const float* codebook = (const float*)d_in[12];
  const float* decW[3]; const float* decB[3];
  for (int i = 0; i < 3; ++i) {
    decW[i] = (const float*)d_in[13 + 2 * i];
    decB[i] = (const float*)d_in[14 + 2 * i];
  }

  float* out = (float*)d_out;
  size_t o_rec = 0;
  size_t o_loss = (size_t)MROWS * 3;
  size_t o_perp = o_loss + 1;
  size_t o_enc = o_perp + 1;
  size_t o_q = o_enc + (size_t)MROWS * ED;
  size_t o_mask = o_q + (size_t)MROWS * ED;
  size_t o_idx = o_mask + MROWS;

  char* ws = (char*)d_ws;
  size_t off = 0;
  auto alloc = [&](size_t bytes) {
    size_t p = off;
    off += (bytes + 255) & ~(size_t)255;
    return p;
  };
  float* xdense = (float*)(ws + alloc((size_t)MROWS * 3 * 4));
  _Float16* h0 = (_Float16*)(ws + alloc((size_t)MROWS * HD * 2));
  _Float16* h1 = (_Float16*)(ws + alloc((size_t)MROWS * HD * 2));
  _Float16* wf[6];
  for (int i = 0; i < 6; ++i) wf[i] = (_Float16*)(ws + alloc((size_t)HD * HD * 2));
  _Float16* cbp = (_Float16*)(ws + alloc((size_t)ED * NEC * 2));
  float* enorm = (float*)(ws + alloc(NEC * 4));
  float* counts = (float*)(ws + alloc(NEC * 4));
  float* sumsq = (float*)(ws + alloc(256));

  hipMemsetAsync(xdense, 0, (size_t)MROWS * 3 * 4, stream);
  hipMemsetAsync(out + o_mask, 0, (size_t)MROWS * 4, stream);
  hipMemsetAsync(counts, 0, NEC * 4, stream);
  hipMemsetAsync(sumsq, 0, 4, stream);

  scatter_kernel<<<(NATOMS + 255) / 256, 256, 0, stream>>>(coords, batch,
                                                           xdense, out + o_mask,
                                                           NATOMS);

  // Pack square weights (K=N=256, row-major KxN: sk=N, sn=1) to fragment form.
  const float* wsrc[6] = {encW[1], encW[2], encW[3], encW[4], decW[0], decW[1]};
  for (int i = 0; i < 6; ++i)
    pack_wfrag_kernel<<<(HD * HD + 255) / 256, 256, 0, stream>>>(
        wsrc[i], wf[i], HD, HD, HD, 1, 1.0f);
  // Codebook: B[k][n] = -2*cb[n][k]  (src [NEC][ED]: sk=1, sn=ED), N=512 K=256
  pack_wfrag_kernel<<<(ED * NEC + 255) / 256, 256, 0, stream>>>(
      codebook, cbp, NEC, ED, 1, ED, -2.0f);
  enorm_kernel<<<(NEC + 63) / 64, 64, 0, stream>>>(codebook, enorm);

  enc_l0_kernel<<<(MROWS * HD) / 256, 256, 0, stream>>>(xdense, encW[0],
                                                        encB[0], h0);

  dim3 g(MROWS / 32, HD / 64);
  gemm16_kernel<1, true, false><<<g, 32, 0, stream>>>(h0, wf[0], encB[1], HD,
                                                      HD, h1, nullptr);
  gemm16_kernel<1, true, false><<<g, 32, 0, stream>>>(h1, wf[1], encB[2], HD,
                                                      HD, h0, nullptr);
  gemm16_kernel<1, true, false><<<g, 32, 0, stream>>>(h0, wf[2], encB[3], HD,
                                                      HD, h1, nullptr);
  // final encoder layer: clip(-10,10); f32 "encoded" output + f16 copy for VQ
  gemm16_kernel<2, true, true><<<g, 32, 0, stream>>>(h1, wf[3], encB[4], ED,
                                                     HD, h0, out + o_enc);

  vq_kernel<<<MROWS / 16, 32, 0, stream>>>(h0, cbp, enorm, codebook,
                                           out + o_mask, out + o_q, h1,
                                           out + o_idx, sumsq, counts);

  gemm16_kernel<1, true, false><<<g, 32, 0, stream>>>(h1, wf[4], decB[0], HD,
                                                      ED, h0, nullptr);
  gemm16_kernel<1, true, false><<<g, 32, 0, stream>>>(h0, wf[5], decB[1], HD,
                                                      HD, h1, nullptr);
  dec_out_kernel<<<(MROWS * 3 + 255) / 256, 256, 0, stream>>>(h1, decW[2],
                                                              decB[2],
                                                              out + o_rec);

  finalize_kernel<<<1, NEC, 0, stream>>>(counts, sumsq, out + o_loss,
                                         out + o_perp);
}
